// SAGEConvModel_21981642620996
// MI455X (gfx1250) — matrix operand order repaired and checked by reading, not compiled, not run on hardware
//
#include <hip/hip_runtime.h>
#include <stdint.h>

typedef __bf16 bf16_t;
typedef __bf16 bf16x4 __attribute__((ext_vector_type(4)));
typedef __bf16 bf16x8 __attribute__((ext_vector_type(8)));
typedef __bf16 v16bf  __attribute__((ext_vector_type(16)));
typedef float  v8f    __attribute__((ext_vector_type(8)));
typedef float  f4     __attribute__((ext_vector_type(4)));

#define IN_CH 1024
#define FEAT  128
#define NEG_SLOPE 0.01f

// ---------------- weight prep: pack [W1_l | W1_r] transposed to [n][k] bf16 ----------------
__global__ __launch_bounds__(256) void k_prep_w1(const float* __restrict__ W1l,
                                                 const float* __restrict__ W1r,
                                                 bf16_t* __restrict__ Bt) {
  int idx = blockIdx.x * 256 + threadIdx.x;     // exactly 256*1024 threads
  int n = idx >> 10, k = idx & 1023;
  float v = (n < FEAT) ? W1l[k * FEAT + n] : W1r[k * FEAT + (n - FEAT)];
  Bt[idx] = (bf16_t)v;
}

// pack [W2_l | W2_r | 0-pad] transposed to [16][128] bf16
__global__ __launch_bounds__(256) void k_prep_w2(const float* __restrict__ W2l,
                                                 const float* __restrict__ W2r,
                                                 bf16_t* __restrict__ W2t) {
  int idx = blockIdx.x * 256 + threadIdx.x;     // 16*128 = 2048
  if (idx >= 16 * FEAT) return;
  int n = idx >> 7, k = idx & 127;
  float v = 0.f;
  if (n < 3)      v = W2l[k * 3 + n];
  else if (n < 6) v = W2r[k * 3 + (n - 3)];
  W2t[idx] = (bf16_t)v;
}

__global__ __launch_bounds__(256) void k_zero(float* __restrict__ p, long n) {
  long i = (long)blockIdx.x * 256 + threadIdx.x;
  if (i < n) p[i] = 0.f;
}

__global__ __launch_bounds__(256) void k_degree(const long long* __restrict__ dst,
                                                float* __restrict__ deg, int E) {
  int e = blockIdx.x * 256 + threadIdx.x;
  if (e < E) atomicAdd(&deg[(int)dst[e]], 1.0f);
}

// ---------------- GEMM1: Y[N x 256] = A[N x 1024](f32->bf16) * Bt[256 x 1024]^T ----------------
// 256 threads = 8 waves; block tile 128(M) x 128(N); K-step 32; software-pipelined staging.
__global__ __launch_bounds__(256) void k_gemm1(const float* __restrict__ A,
                                               const bf16_t* __restrict__ Bt,
                                               float* __restrict__ Y, int N) {
  __shared__ __align__(16) bf16_t sA[128 * 32];   // A tile, row-major [m][k]
  __shared__ __align__(16) bf16_t sB[128 * 32];   // B tile, [n][k] (pre-transposed)
  const int tid   = threadIdx.x;
  const int lane  = tid & 31;
  const int w     = tid >> 5;
  const int waveM = w & 3;       // 4 waves along M (32 rows each)
  const int waveN = w >> 2;      // 2 waves along N (64 cols each)
  const int m0    = blockIdx.x * 128;
  const int nBase = blockIdx.y * 128;
  const int sel   = lane >> 4;
  const int l15   = lane & 15;

  // Per-thread staging coordinates (A: 4 float4 slots, B: 2 16B chunks).
  // Row clamp instead of predication: A rows map 1:1 to C rows, so clamped
  // (duplicate) rows only affect never-stored outputs -> no exec divergence,
  // loads batch freely.
  int aRow[4], aKc[4];
  size_t aOff[4];
#pragma unroll
  for (int j = 0; j < 4; ++j) {
    int p = tid + j * 256;               // 0..1023 float4 slots
    aRow[j] = p >> 3;
    aKc[j]  = (p & 7) << 2;
    int m = m0 + aRow[j];
    if (m >= N) m = N - 1;
    aOff[j] = (size_t)m * IN_CH + aKc[j];
  }
  int bN[2], bKk[2];
#pragma unroll
  for (int j = 0; j < 2; ++j) {
    int p = tid + j * 256;               // 0..511 8-elt chunks
    bN[j]  = p >> 2;
    bKk[j] = (p & 3) << 3;
  }

  f4     aReg[4];
  bf16x8 bReg[2];

  const v8f vzero = {0.f, 0.f, 0.f, 0.f, 0.f, 0.f, 0.f, 0.f};
  v8f acc[2][4];
#pragma unroll
  for (int s = 0; s < 2; ++s)
#pragma unroll
    for (int t = 0; t < 4; ++t) acc[s][t] = vzero;

  // prologue fetch (k0 = 0)
#pragma unroll
  for (int j = 0; j < 4; ++j)
    aReg[j] = __builtin_nontemporal_load((const f4*)(A + aOff[j]));   // stream-once
#pragma unroll
  for (int j = 0; j < 2; ++j)
    bReg[j] = *(const bf16x8*)(Bt + (size_t)(nBase + bN[j]) * IN_CH + bKk[j]);

  for (int kt = 0; kt < IN_CH / 32; ++kt) {
    __syncthreads();                     // previous tile's frag reads complete
    // commit staged registers to LDS (packed f32->bf16 converts, vector stores)
#pragma unroll
    for (int j = 0; j < 4; ++j)
      *(bf16x4*)&sA[aRow[j] * 32 + aKc[j]] = __builtin_convertvector(aReg[j], bf16x4);
#pragma unroll
    for (int j = 0; j < 2; ++j)
      *(bf16x8*)&sB[bN[j] * 32 + bKk[j]] = bReg[j];
    __syncthreads();

    // issue next tile's global loads; s_wait_loadcnt lands at next commit,
    // hidden under the 8 WMMAs below
    if (kt + 1 < IN_CH / 32) {
      const int k0n = (kt + 1) * 32;
#pragma unroll
      for (int j = 0; j < 4; ++j)
        aReg[j] = __builtin_nontemporal_load((const f4*)(A + aOff[j] + k0n));
#pragma unroll
      for (int j = 0; j < 2; ++j)
        bReg[j] = *(const bf16x8*)(Bt + (size_t)(nBase + bN[j]) * IN_CH + k0n + bKk[j]);
    }

    // fragments per ISA 16-bit layouts
    v16bf af[2], bfr[4];
#pragma unroll
    for (int s = 0; s < 2; ++s) {
      int r = waveM * 32 + s * 16 + l15;
      const bf16_t* base = &sA[r * 32 + sel * 8];       // K = sel*8..+7, 16+sel*8..+7
      bf16x8 lo = *(const bf16x8*)base;
      bf16x8 hi = *(const bf16x8*)(base + 16);
      af[s] = __builtin_shufflevector(lo, hi, 0,1,2,3,4,5,6,7,8,9,10,11,12,13,14,15);
    }
#pragma unroll
    for (int t = 0; t < 4; ++t) {
      int c = waveN * 64 + t * 16 + l15;
      const bf16_t* base = &sB[c * 32 + sel * 8];       // K = sel*8..+7, 16+sel*8..+7
      bf16x8 lo = *(const bf16x8*)base;
      bf16x8 hi = *(const bf16x8*)(base + 16);
      bfr[t] = __builtin_shufflevector(lo, hi, 0,1,2,3,4,5,6,7,8,9,10,11,12,13,14,15);
    }
#pragma unroll
    for (int s = 0; s < 2; ++s)
#pragma unroll
      for (int t = 0; t < 4; ++t)
        acc[s][t] = __builtin_amdgcn_wmma_f32_16x16x32_bf16(
            false, af[s], false, bfr[t], (short)0, acc[s][t], false, false);
  }

  const int ro = sel * 8;
#pragma unroll
  for (int s = 0; s < 2; ++s)
#pragma unroll
    for (int t = 0; t < 4; ++t) {
      int n = nBase + waveN * 64 + t * 16 + l15;
#pragma unroll
      for (int r = 0; r < 8; ++r) {
        int m = m0 + waveM * 32 + s * 16 + ro + r;
        if (m < N) Y[(size_t)m * 256 + n] = acc[s][t][r];
      }
    }
}

// ---------------- edge aggregation in 128-dim (post-GEMM, 8x less traffic) ----------------
__global__ __launch_bounds__(256) void k_agg1(const float* __restrict__ Y,
                                              const long long* __restrict__ src,
                                              const long long* __restrict__ dst,
                                              float* __restrict__ agg, int E) {
  int gid = blockIdx.x * 256 + threadIdx.x;
  int e = gid >> 5;                  // 32 lanes per edge, 4 channels/lane
  if (e >= E) return;
  int lane = gid & 31;
  int s = (int)src[e], d = (int)dst[e];
  f4 v = *(const f4*)(Y + (size_t)s * 256 + lane * 4);   // cols 0..127 = Y_l
  float* o = agg + (size_t)d * FEAT + lane * 4;
  atomicAdd(o + 0, v.x); atomicAdd(o + 1, v.y);
  atomicAdd(o + 2, v.z); atomicAdd(o + 3, v.w);
}

// x1 = leaky_relu(agg/deg + Y_r + b1), in place over agg
__global__ __launch_bounds__(256) void k_combine1(float* __restrict__ agg_x1,
                                                  const float* __restrict__ deg,
                                                  const float* __restrict__ Y,
                                                  const float* __restrict__ b1, int N) {
  int idx = blockIdx.x * 256 + threadIdx.x;
  if (idx >= N * FEAT) return;
  int m = idx >> 7, c = idx & 127;
  float inv = 1.f / fmaxf(deg[m], 1.f);
  float x = agg_x1[idx] * inv + Y[(size_t)m * 256 + FEAT + c] + b1[c];
  agg_x1[idx] = (x >= 0.f) ? x : NEG_SLOPE * x;
}

// ---------------- GEMM2: Z[N x 16] = x1[N x 128] * W2t[16 x 128]^T ----------------
__global__ __launch_bounds__(256) void k_gemm2(const float* __restrict__ X,
                                               const bf16_t* __restrict__ W2t,
                                               float* __restrict__ Z, int N) {
  __shared__ __align__(16) bf16_t sX[128 * 128];   // 32 KB
  __shared__ __align__(16) bf16_t sW[16 * 128];    // 4 KB
  const int tid  = threadIdx.x;
  const int lane = tid & 31;
  const int w    = tid >> 5;        // 8 waves, 16 rows each
  const int m0   = blockIdx.x * 128;
  const int sel  = lane >> 4;
  const int l15  = lane & 15;

#pragma unroll
  for (int j = 0; j < 16; ++j) {
    int p = tid + j * 256;          // 0..4095 float4 slots
    int row = p >> 5;
    int kc  = (p & 31) << 2;
    int m = m0 + row;
    if (m >= N) m = N - 1;          // clamp: duplicate rows never stored
    f4 v = *(const f4*)(X + (size_t)m * FEAT + kc);
    *(bf16x4*)&sX[row * FEAT + kc] = __builtin_convertvector(v, bf16x4);
  }
  {
    int n  = tid >> 4;
    int kk = (tid & 15) << 3;
    *(bf16x8*)&sW[n * FEAT + kk] = *(const bf16x8*)(W2t + n * FEAT + kk);
  }
  __syncthreads();

  v8f acc = {0.f, 0.f, 0.f, 0.f, 0.f, 0.f, 0.f, 0.f};
#pragma unroll
  for (int kt = 0; kt < FEAT / 32; ++kt) {
    int k0 = kt * 32;
    int r = w * 16 + l15;
    const bf16_t* ab = &sX[r * FEAT + k0 + sel * 8];
    bf16x8 alo = *(const bf16x8*)ab;
    bf16x8 ahi = *(const bf16x8*)(ab + 16);
    v16bf af = __builtin_shufflevector(alo, ahi, 0,1,2,3,4,5,6,7,8,9,10,11,12,13,14,15);
    const bf16_t* bb = &sW[l15 * FEAT + k0 + sel * 8];
    bf16x8 blo = *(const bf16x8*)bb;
    bf16x8 bhi = *(const bf16x8*)(bb + 16);
    v16bf bf_ = __builtin_shufflevector(blo, bhi, 0,1,2,3,4,5,6,7,8,9,10,11,12,13,14,15);
    acc = __builtin_amdgcn_wmma_f32_16x16x32_bf16(false, af, false, bf_, (short)0, acc,
                                                  false, false);
  }
  int ro = sel * 8;
#pragma unroll
  for (int r = 0; r < 8; ++r) {
    int m = m0 + w * 16 + ro + r;
    if (m < N) Z[(size_t)m * 16 + l15] = acc[r];
  }
}

__global__ __launch_bounds__(256) void k_agg2(const float* __restrict__ Z,
                                              const long long* __restrict__ src,
                                              const long long* __restrict__ dst,
                                              float* __restrict__ aggZ, int E) {
  int e = blockIdx.x * 256 + threadIdx.x;
  if (e >= E) return;
  int s = (int)src[e], d = (int)dst[e];
  const float* z = Z + (size_t)s * 16;          // cols 0..2 = Z_l
  atomicAdd(&aggZ[(size_t)d * 4 + 0], z[0]);
  atomicAdd(&aggZ[(size_t)d * 4 + 1], z[1]);
  atomicAdd(&aggZ[(size_t)d * 4 + 2], z[2]);
}

__global__ __launch_bounds__(256) void k_final(const float* __restrict__ aggZ,
                                               const float* __restrict__ deg,
                                               const float* __restrict__ Z,
                                               const float* __restrict__ b2,
                                               float* __restrict__ out, int N) {
  int idx = blockIdx.x * 256 + threadIdx.x;
  if (idx >= N * 4) return;
  int m = idx >> 2, c = idx & 3;
  if (c >= 3) return;
  float inv = 1.f / fmaxf(deg[m], 1.f);
  out[m * 3 + c] = aggZ[idx] * inv + Z[(size_t)m * 16 + 3 + c] + b2[c];
}

extern "C" void kernel_launch(void* const* d_in, const int* in_sizes, int n_in,
                              void* d_out, int out_size, void* d_ws, size_t ws_size,
                              hipStream_t stream) {
  (void)n_in; (void)out_size; (void)ws_size;
  const float*     feats = (const float*)d_in[0];
  const long long* e2    = (const long long*)d_in[2];   // edges2, jnp.int64
  const float*     W1l   = (const float*)d_in[5];
  const float*     W1r   = (const float*)d_in[6];
  const float*     b1    = (const float*)d_in[7];
  const float*     W2l   = (const float*)d_in[8];
  const float*     W2r   = (const float*)d_in[9];
  const float*     b2    = (const float*)d_in[10];
  float*           out   = (float*)d_out;

  const int N = in_sizes[0] / IN_CH;
  const int E = in_sizes[2] / 2;
  const long long* src = e2;
  const long long* dst = e2 + E;

  char* ws = (char*)d_ws;
  size_t off = 0;
  auto take = [&](size_t bytes) -> void* {
    void* p = ws + off;
    off += (bytes + 255) & ~(size_t)255;
    return p;
  };
  bf16_t* Bt1  = (bf16_t*)take((size_t)256 * IN_CH * sizeof(bf16_t));
  bf16_t* W2t  = (bf16_t*)take((size_t)16 * FEAT * sizeof(bf16_t));
  float*  Y    = (float*)take((size_t)N * 256 * sizeof(float));
  float*  agg  = (float*)take((size_t)N * FEAT * sizeof(float));  // becomes x1
  float*  deg  = (float*)take((size_t)N * sizeof(float));
  float*  Z    = (float*)take((size_t)N * 16 * sizeof(float));
  float*  aggZ = (float*)take((size_t)N * 4 * sizeof(float));

  k_prep_w1<<<(256 * IN_CH) / 256, 256, 0, stream>>>(W1l, W1r, Bt1);
  k_prep_w2<<<(16 * FEAT + 255) / 256, 256, 0, stream>>>(W2l, W2r, W2t);
  k_zero<<<(int)(((long)N * FEAT + 255) / 256), 256, 0, stream>>>(agg, (long)N * FEAT);
  k_zero<<<(N + 255) / 256, 256, 0, stream>>>(deg, (long)N);
  k_zero<<<(int)(((long)N * 4 + 255) / 256), 256, 0, stream>>>(aggZ, (long)N * 4);
  k_degree<<<(E + 255) / 256, 256, 0, stream>>>(dst, deg, E);

  dim3 g1((N + 127) / 128, 2);
  k_gemm1<<<g1, 256, 0, stream>>>(feats, Bt1, Y, N);

  k_agg1<<<(int)(((long)E * 32 + 255) / 256), 256, 0, stream>>>(Y, src, dst, agg, E);
  k_combine1<<<(int)(((long)N * FEAT + 255) / 256), 256, 0, stream>>>(agg, deg, Y, b1, N);

  k_gemm2<<<(N + 127) / 128, 256, 0, stream>>>(agg, W2t, Z, N);
  k_agg2<<<(E + 255) / 256, 256, 0, stream>>>(Z, src, dst, aggZ, E);
  k_final<<<(int)(((long)N * 4 + 255) / 256), 256, 0, stream>>>(aggZ, deg, Z, b2, out, N);
}
